// Hybrid_Python_76527727280368
// MI455X (gfx1250) — matrix operand order repaired and checked by reading, not compiled, not run on hardware
//
#include <hip/hip_runtime.h>
#include <math.h>

#define D_DIM 2048
#define H_DIM 8192
#define T_STEPS 4096
#define NWG 128
#define NTHREADS 256
#define NWAVES 8

#define KT1 (D_DIM / 32)   // 64  K-tiles per column group in W1
#define KT2 (H_DIM / 32)   // 256 K-tiles per column group in W2

typedef __attribute__((ext_vector_type(2)))  float  v2f;
typedef __attribute__((ext_vector_type(8)))  float  v8f;
typedef __attribute__((ext_vector_type(8)))  __bf16 v8bf;
typedef __attribute__((ext_vector_type(16))) __bf16 v16bf;

// ---------------------------------------------------------------------------
// Grid-wide barrier: monotone atomic counter, acquire spin with s_sleep.
// ---------------------------------------------------------------------------
__device__ __forceinline__ void grid_barrier(unsigned* cnt, unsigned expected) {
  __syncthreads();
  if (threadIdx.x == 0) {
    __hip_atomic_fetch_add(cnt, 1u, __ATOMIC_ACQ_REL, __HIP_MEMORY_SCOPE_AGENT);
    while (__hip_atomic_load(cnt, __ATOMIC_ACQUIRE, __HIP_MEMORY_SCOPE_AGENT) < expected) {
      __builtin_amdgcn_s_sleep(2);
    }
  }
  __syncthreads();
}

__global__ void init_kernel(const float* __restrict__ x0, float* __restrict__ out,
                            float* __restrict__ xg, unsigned* __restrict__ bar) {
  int i = blockIdx.x * blockDim.x + threadIdx.x;
  if (i == 0) *bar = 0u;
  if (i < D_DIM) {
    float v = x0[i];
    out[i] = v;
    xg[i]  = v;
  }
}

// ---------------------------------------------------------------------------
// Pack W (f32, Ktot x N row-major) into bf16 WMMA-B tiles.
// Tile (nt, kt) covers K rows kt*32..+32, N cols nt*16..+16; tiles stored
// K-major within an N-group: tileIdx = nt*(Ktot/32) + kt, 512 elems (1 KB).
// In-tile: lane L (0..31) owns 16 contiguous bf16 = column N = nt*16 + (L&15),
// K = kt*32 + (j>>3)*16 + (L>>4)*8 + (j&7)  (j = 0..15)  -> 16-bit B 32x16 layout
// (lanes 0-15: K=0..7,16..23 as pairs per VGPR, lanes 16-31: K=8..15,24..31).
// ---------------------------------------------------------------------------
__global__ void pack_bf16_kernel(const float* __restrict__ W, __bf16* __restrict__ P,
                                 int Ktot, int N) {
  size_t e = (size_t)blockIdx.x * blockDim.x + threadIdx.x;
  size_t total = (size_t)Ktot * (size_t)N;
  if (e >= total) return;
  const int KT = Ktot >> 5;
  size_t tile = e >> 9;
  int L  = (int)((e >> 4) & 31);
  int j  = (int)(e & 15);
  int nt = (int)(tile / (size_t)KT);
  int kt = (int)(tile % (size_t)KT);
  int k  = kt * 32 + (j >> 3) * 16 + (L >> 4) * 8 + (j & 7);
  int n  = nt * 16 + (L & 15);
  P[e] = (__bf16)W[(size_t)k * (size_t)N + n];
}

// ---------------------------------------------------------------------------
// bf16 GEMV tile: partial y[nt*16 .. +16) over K-tiles [kt0, kt0+ktn).
// A = x broadcast over all 16 M rows (every C row identical; result for
// column n is c[0] of lane n). A built from bf16 x in LDS per the documented
// 16-bit A 16x32 layout: lane half K offset {0 | 8}, chunks [0..7] & [16..23].
// Unroll capped at 4 so at most ~8 B tiles are in flight -> no VGPR spills.
// ---------------------------------------------------------------------------
__device__ __forceinline__ v8f gemv_bf16_tile(const __bf16* __restrict__ P,
                                              const __bf16* xbl,
                                              int KT, int nt, int kt0, int ktn,
                                              int lane) {
  v8f c0 = {};
  v8f c1 = {};
  const int lo = (lane >= 16) ? 8 : 0;
  const __bf16* wp = P + (((size_t)nt * (size_t)KT + (size_t)kt0) << 9) + lane * 16;
  const __bf16* xp = xbl + kt0 * 32 + lo;
#pragma clang loop unroll_count(4)
  for (int i = 0; i < ktn; i += 2) {
    __builtin_prefetch(wp + 2048, 0, 3);     // global_prefetch_b8, 2 iters ahead
    v8bf a0lo = *(const v8bf*)(xp);
    v8bf a0hi = *(const v8bf*)(xp + 16);
    v8bf a1lo = *(const v8bf*)(xp + 32);
    v8bf a1hi = *(const v8bf*)(xp + 48);
    v16bf b0  = *(const v16bf*)(wp);
    v16bf b1  = *(const v16bf*)(wp + 512);
    v16bf a0  = __builtin_shufflevector(a0lo, a0hi, 0, 1, 2, 3, 4, 5, 6, 7,
                                        8, 9, 10, 11, 12, 13, 14, 15);
    v16bf a1  = __builtin_shufflevector(a1lo, a1hi, 0, 1, 2, 3, 4, 5, 6, 7,
                                        8, 9, 10, 11, 12, 13, 14, 15);
    c0 = __builtin_amdgcn_wmma_f32_16x16x32_bf16(false, a0, false, b0,
                                                 (short)0, c0, false, false);
    c1 = __builtin_amdgcn_wmma_f32_16x16x32_bf16(false, a1, false, b1,
                                                 (short)0, c1, false, false);
    xp += 64;
    wp += 1024;
  }
  return c0 + c1;
}

// ---------------------------------------------------------------------------
// bf16 persistent Euler kernel (primary path).
// ---------------------------------------------------------------------------
__global__ void __launch_bounds__(NTHREADS)
hybrid_euler_bf16(const float* __restrict__ t,
                  const __bf16* __restrict__ W1p, const float* __restrict__ b1,
                  const __bf16* __restrict__ W2p, const float* __restrict__ b2,
                  const float* __restrict__ damping,
                  float* __restrict__ out,
                  unsigned* __restrict__ bar,
                  float* __restrict__ xg, float* __restrict__ hg) {
  __shared__ __bf16 xbl[D_DIM];          // state (bf16) for A operand
  __shared__ __bf16 hbl[H_DIM];          // hidden (bf16) for A operand
  __shared__ float  red[NWAVES][16];

  const int tid  = threadIdx.x;
  const int lane = tid & 31;
  const int wave = tid >> 5;
  const int wg   = blockIdx.x;
  unsigned bar_target = 0;

  for (int step = 0; step < T_STEPS - 1; ++step) {
    for (int i = tid; i < D_DIM; i += NTHREADS) xbl[i] = (__bf16)xg[i];
    __syncthreads();

    // ---- Phase 1: h = tanh(x @ W1 + b1); 512 tiles, 2 waves/tile ----
    {
      const int tile  = wg * 4 + (wave >> 1);
      const int khalf = wave & 1;
      v8f c = gemv_bf16_tile(W1p, xbl, KT1, tile, khalf * (KT1 / 2), KT1 / 2, lane);
      if (lane < 16) red[wave][lane] = c[0];
      __syncthreads();
      if (tid < 64) {
        const int tl = tid >> 4;
        const int n  = tid & 15;
        const int ng = (wg * 4 + tl) * 16 + n;
        float s = red[2 * tl][n] + red[2 * tl + 1][n];
        hg[ng] = tanhf(s + b1[ng]);
      }
    }
    bar_target += NWG;
    grid_barrier(bar, bar_target);

    for (int i = tid; i < H_DIM; i += NTHREADS) hbl[i] = (__bf16)hg[i];
    __syncthreads();

    // ---- Phase 2: y = h @ W2; one tile per WG, 8-way K split ----
    {
      const int nt = wg;
      v8f c = gemv_bf16_tile(W2p, hbl, KT2, nt, wave * (KT2 / 8), KT2 / 8, lane);
      __syncthreads();
      if (lane < 16) red[wave][lane] = c[0];
      __syncthreads();
      if (tid < 16) {
        float s = 0.f;
#pragma unroll
        for (int wv = 0; wv < NWAVES; ++wv) s += red[wv][tid];
        const int d = nt * 16 + tid;
        const float y   = s + b2[d];
        const float xo  = xg[d];                 // fp32 state for accuracy
        const float dtv = t[step + 1] - t[step];
        const float xn  = xo + dtv * (y - damping[d] * xo);
        out[(size_t)(step + 1) * D_DIM + d] = xn;
        xg[d] = xn;
      }
    }
    bar_target += NWG;
    grid_barrier(bar, bar_target);
  }
}

// ---------------------------------------------------------------------------
// f32 WMMA fallback (used when ws_size cannot hold packed bf16 weights).
// ---------------------------------------------------------------------------
__device__ __forceinline__ v8f gemv_f32_tile(const float* __restrict__ W,
                                             const float* __restrict__ xl,
                                             int ldW, int n0, int k0, int kcnt,
                                             int lane) {
  v8f c0 = {};
  v8f c1 = {};
  const int koff = (lane >= 16) ? 2 : 0;
  const int col  = lane & 15;
  const float* xp = xl + k0 + koff;
  const float* wp = W + (size_t)(k0 + koff) * (size_t)ldW + (size_t)n0 + (size_t)col;
  const size_t ld = (size_t)ldW;
#pragma clang loop unroll_count(4)
  for (int k = 0; k < kcnt; k += 8) {
    v2f a0, b0, a1, b1;
    a0.x = xp[0];      a0.y = xp[1];
    b0.x = wp[0];      b0.y = wp[ld];
    a1.x = xp[4];      a1.y = xp[5];
    b1.x = wp[4 * ld]; b1.y = wp[5 * ld];
    c0 = __builtin_amdgcn_wmma_f32_16x16x4_f32(false, a0, false, b0, (short)0, c0, false, false);
    c1 = __builtin_amdgcn_wmma_f32_16x16x4_f32(false, a1, false, b1, (short)0, c1, false, false);
    xp += 8;
    wp += 8 * ld;
  }
  return c0 + c1;
}

__global__ void __launch_bounds__(NTHREADS)
hybrid_euler_f32(const float* __restrict__ t,
                 const float* __restrict__ W1, const float* __restrict__ b1,
                 const float* __restrict__ W2, const float* __restrict__ b2,
                 const float* __restrict__ damping,
                 float* __restrict__ out,
                 unsigned* __restrict__ bar,
                 float* __restrict__ xg, float* __restrict__ hg) {
  __shared__ float xl[D_DIM];
  __shared__ float hl[H_DIM];
  __shared__ float red[NWAVES][16];

  const int tid  = threadIdx.x;
  const int lane = tid & 31;
  const int wave = tid >> 5;
  const int wg   = blockIdx.x;
  unsigned bar_target = 0;

  for (int step = 0; step < T_STEPS - 1; ++step) {
    for (int i = tid; i < D_DIM; i += NTHREADS) xl[i] = xg[i];
    __syncthreads();
    {
      const int tile  = wg * 4 + (wave >> 1);
      const int khalf = wave & 1;
      v8f c = gemv_f32_tile(W1, xl, H_DIM, tile * 16, khalf * (D_DIM / 2), D_DIM / 2, lane);
      if (lane < 16) red[wave][lane] = c[0];
      __syncthreads();
      if (tid < 64) {
        const int tl = tid >> 4;
        const int n  = tid & 15;
        const int ng = (wg * 4 + tl) * 16 + n;
        float s = red[2 * tl][n] + red[2 * tl + 1][n];
        hg[ng] = tanhf(s + b1[ng]);
      }
    }
    bar_target += NWG;
    grid_barrier(bar, bar_target);

    for (int i = tid; i < H_DIM; i += NTHREADS) hl[i] = hg[i];
    __syncthreads();
    {
      const int n0 = wg * 16;
      v8f c = gemv_f32_tile(W2, hl, D_DIM, n0, wave * (H_DIM / 8), H_DIM / 8, lane);
      __syncthreads();
      if (lane < 16) red[wave][lane] = c[0];
      __syncthreads();
      if (tid < 16) {
        float s = 0.f;
#pragma unroll
        for (int wv = 0; wv < NWAVES; ++wv) s += red[wv][tid];
        const int d = n0 + tid;
        const float y   = s + b2[d];
        const float xo  = xl[d];
        const float dtv = t[step + 1] - t[step];
        const float xn  = xo + dtv * (y - damping[d] * xo);
        out[(size_t)(step + 1) * D_DIM + d] = xn;
        xg[d] = xn;
      }
    }
    bar_target += NWG;
    grid_barrier(bar, bar_target);
  }
}

// ---------------------------------------------------------------------------
extern "C" void kernel_launch(void* const* d_in, const int* in_sizes, int n_in,
                              void* d_out, int out_size, void* d_ws, size_t ws_size,
                              hipStream_t stream) {
  const float* x0      = (const float*)d_in[0];
  const float* t       = (const float*)d_in[1];
  const float* W1      = (const float*)d_in[2];
  const float* b1      = (const float*)d_in[3];
  const float* W2      = (const float*)d_in[4];
  const float* b2      = (const float*)d_in[5];
  const float* damping = (const float*)d_in[6];
  float* out = (float*)d_out;

  char* ws = (char*)d_ws;
  unsigned* bar = (unsigned*)ws;                       // [0, 256): barrier counter
  float* xg = (float*)(ws + 256);                      // fp32 state (8 KB)
  float* hg = (float*)(ws + 256 + D_DIM * 4);          // fp32 hidden (32 KB)

  const size_t W_ELEMS  = (size_t)D_DIM * (size_t)H_DIM;   // 16.8M per matrix
  const size_t PACK_OFF = 65536;                            // header region
  const size_t NEED     = PACK_OFF + 2 * W_ELEMS * 2;       // ~64.06 MB

  init_kernel<<<(D_DIM + 255) / 256, 256, 0, stream>>>(x0, out, xg, bar);

  if (ws_size >= NEED) {
    __bf16* W1p = (__bf16*)(void*)(ws + PACK_OFF);
    __bf16* W2p = (__bf16*)(void*)(ws + PACK_OFF + W_ELEMS * 2);
    const int pack_blocks = (int)((W_ELEMS + 255) / 256);
    pack_bf16_kernel<<<pack_blocks, 256, 0, stream>>>(W1, W1p, D_DIM, H_DIM);
    pack_bf16_kernel<<<pack_blocks, 256, 0, stream>>>(W2, W2p, H_DIM, D_DIM);
    hybrid_euler_bf16<<<NWG, NTHREADS, 0, stream>>>(t, W1p, b1, W2p, b2, damping,
                                                    out, bar, xg, hg);
  } else {
    hybrid_euler_f32<<<NWG, NTHREADS, 0, stream>>>(t, W1, b1, W2, b2, damping,
                                                   out, bar, xg, hg);
  }
}
